// HybridQLSTM_65481071403442
// MI455X (gfx1250) — matrix operand order repaired and checked by reading, not compiled, not run on hardware
//
#include <hip/hip_runtime.h>
#include <hip/hip_bf16.h>
#include <stdint.h>

#define T_STEPS 512
#define BATCH   128
#define DIN     512
#define HID     512
#define DCOMB   1024   // DIN + HID
#define NGATES  2048   // 4*HID
#define NBLK    64     // persistent grid: 64 blocks x 256 threads

typedef __attribute__((ext_vector_type(16))) __bf16 v16bf;
typedef __attribute__((ext_vector_type(8)))  float  v8f;
typedef int v4i_g __attribute__((vector_size(16)));   // matches async builtin param

#if defined(__has_builtin)
#  if __has_builtin(__builtin_amdgcn_global_load_async_to_lds_b128)
#    define HAVE_ASYNC_B128 1
#  endif
#  if __has_builtin(__builtin_amdgcn_s_wait_asynccnt)
#    define HAVE_WAIT_ASYNC 1
#  endif
#endif

__device__ __forceinline__ void async_copy_b128(const void* gsrc, void* lds_dst) {
#if defined(HAVE_ASYNC_B128)
    __builtin_amdgcn_global_load_async_to_lds_b128(
        (__attribute__((address_space(1))) v4i_g*)gsrc,
        (__attribute__((address_space(3))) v4i_g*)lds_dst,
        0, 0);
#else
    unsigned lofs = (unsigned)(uintptr_t)(__attribute__((address_space(3))) void*)lds_dst;
    unsigned long long ga = (unsigned long long)(uintptr_t)gsrc;
    asm volatile("global_load_async_to_lds_b128 %0, %1, off"
                 :: "v"(lofs), "v"(ga) : "memory");
#endif
}

__device__ __forceinline__ void wait_async0() {
#if defined(HAVE_WAIT_ASYNC)
    __builtin_amdgcn_s_wait_asynccnt(0);
#else
    asm volatile("s_wait_asynccnt 0x0" ::: "memory");
#endif
}

__device__ __forceinline__ unsigned short f32_to_bf16_rne(float f) {
    union { float f; uint32_t u; } cv; cv.f = f;
    uint32_t u = cv.u;
    uint32_t rounded = u + 0x7FFFu + ((u >> 16) & 1u);
    rounded = ((u & 0x7F800000u) == 0x7F800000u) ? u : rounded;
    return (unsigned short)(rounded >> 16);
}

// device-wide barrier: monotonic counter, per-block epoch target (no reset races)
__device__ __forceinline__ void grid_sync(unsigned* cnt, unsigned target) {
    __threadfence();      // each wave: own global stores visible at device scope
    __syncthreads();
    if (threadIdx.x == 0) {
        __hip_atomic_fetch_add(cnt, 1u, __ATOMIC_ACQ_REL, __HIP_MEMORY_SCOPE_AGENT);
        while (__hip_atomic_load(cnt, __ATOMIC_ACQUIRE, __HIP_MEMORY_SCOPE_AGENT) < target)
            __builtin_amdgcn_s_sleep(2);
    }
    __syncthreads();
}

// ---------------------------------------------------------------------------
// Prep: swizzle W = concat(Wf,Wi,Wg,Wo) [1024 x 2048] f32 into bf16 tiles in
// a contiguous-k (k = 16h + p) B-matrix lane order. Tile (nt,kt) is 512 bf16 at
// base ((nt*32)+kt)*512; element lane*16+p = W[kt*32 + (lane<16?0:16)+p][nt*16+lane%16]
// ---------------------------------------------------------------------------
__global__ void prep_w_kernel(const float* __restrict__ Wf, const float* __restrict__ Wi,
                              const float* __restrict__ Wg, const float* __restrict__ Wo,
                              unsigned short* __restrict__ Wsw) {
    int e = blockIdx.x * blockDim.x + threadIdx.x;
    if (e >= DCOMB * NGATES) return;
    int tile = e >> 9;
    int r    = e & 511;
    int lane = r >> 4;
    int p    = r & 15;
    int nt   = tile >> 5;
    int kt   = tile & 31;
    int row  = kt * 32 + ((lane < 16) ? 0 : 16) + p;   // 0..1023
    int col  = nt * 16 + (lane & 15);                  // 0..2047
    int g    = col >> 9;
    int cc   = col & 511;
    const float* Wsrc = (g == 0) ? Wf : (g == 1) ? Wi : (g == 2) ? Wg : Wo;
    Wsw[e] = f32_to_bf16_rne(Wsrc[row * HID + cc]);
}

// bias pack + state init + convert x_0 + zero the grid-sync counter
__global__ void prep_state_kernel(const float* __restrict__ x0,
                                  const float* __restrict__ bf, const float* __restrict__ bi,
                                  const float* __restrict__ bg, const float* __restrict__ bo,
                                  float* __restrict__ bias, unsigned short* __restrict__ comb,
                                  float* __restrict__ c_state, unsigned* __restrict__ sync_cnt) {
    int i = blockIdx.x * blockDim.x + threadIdx.x;
    if (i == 0) *sync_cnt = 0u;
    if (i < NGATES) {
        int g = i >> 9, cc = i & 511;
        const float* src = (g == 0) ? bf : (g == 1) ? bi : (g == 2) ? bg : bo;
        bias[i] = src[cc];
    }
    if (i < BATCH * HID) {
        int b = i >> 9, j = i & 511;
        comb[(size_t)b * DCOMB + j]       = f32_to_bf16_rne(x0[i]); // x half
        comb[(size_t)b * DCOMB + HID + j] = 0;                      // h half = 0
        c_state[i] = 0.0f;
    }
}

// ---------------------------------------------------------------------------
// Persistent LSTM: 64 blocks x 256 threads loop over all 512 timesteps.
// Each block owns 32 N-columns; its W slab (32 cols x 1024 K bf16 = 64KB)
// is staged into LDS ONCE (async TDM-class copies) and stays resident.
// Per step: barrier-free WMMA K-loop -> gates; grid sync; fused pointwise
// LSTM update (+ h->bf16 and x_{t+1}->bf16 into the combined A buffer);
// grid sync.
// ---------------------------------------------------------------------------
__global__ __launch_bounds__(256)
void lstm_persistent_kernel(const float* __restrict__ inputs,   // [T, B, DIN] f32
                            unsigned short* __restrict__ comb,  // [B, DCOMB] bf16 (x|h)
                            const unsigned short* __restrict__ Wsw,
                            const float* __restrict__ bias,
                            float* __restrict__ gates,          // [B, NGATES] f32
                            float* __restrict__ c_state,        // [B, HID] f32
                            float* __restrict__ out,            // outputs | hx | cx
                            unsigned* __restrict__ sync_cnt)
{
    __shared__ __align__(16) unsigned short lsW[2 * 32 * 512];  // 64KB resident W slab

    const int tid     = threadIdx.x;
    const int wave    = tid >> 5;
    const int lane    = tid & 31;
    const int nregion = blockIdx.x * 32;   // this block's 32 N-columns
    const int ntbase  = blockIdx.x * 2;    // first of its 2 N-tiles
    const int mrow    = wave * 16 + (lane & 15);
    const int halfsel = lane >> 4;         // 0 or 1

    // one-time async stage of the whole W slab (16 x b128 per thread)
    {
        const unsigned short* src = Wsw + (size_t)ntbase * 32 * 512;
#pragma unroll
        for (int s = 0; s < 16; ++s) {
            int eo = s * 2048 + tid * 8;
            async_copy_b128(src + eo, &lsW[eo]);
        }
        wait_async0();
        __syncthreads();
    }

    const unsigned short* arow = comb + (size_t)mrow * DCOMB;
    float bn0 = bias[nregion + (lane & 15)];
    float bn1 = bias[nregion + 16 + (lane & 15)];

    unsigned epoch = 0;

    for (int t = 0; t < T_STEPS; ++t) {
        // ---------------- GEMM phase (no barriers in the K loop) ------------
        v8f acc0, acc1;
#pragma unroll
        for (int v = 0; v < 8; ++v) { acc0[v] = 0.0f; acc1[v] = 0.0f; }

        for (int kt = 0; kt < 32; ++kt) {
            // A fragment: two contiguous 16B runs of the combined [x|h] row
            union { v16bf v; uint4 q[2]; } afrag;
            const int k0 = kt * 32 + halfsel * 16;
            afrag.q[0] = *(const uint4*)(arow + k0);
            afrag.q[1] = *(const uint4*)(arow + k0 + 8);

            // B fragments straight from the resident LDS slab
            union { v16bf v; uint4 u[2]; } b0, b1;
            const uint4* bp0 = (const uint4*)(lsW + (size_t)kt * 512 + lane * 16);
            const uint4* bp1 = (const uint4*)(lsW + (size_t)(32 + kt) * 512 + lane * 16);
            b0.u[0] = bp0[0]; b0.u[1] = bp0[1];
            b1.u[0] = bp1[0]; b1.u[1] = bp1[1];

            acc0 = __builtin_amdgcn_wmma_f32_16x16x32_bf16(
                false, afrag.v, false, b0.v, (short)0, acc0, false, false);
            acc1 = __builtin_amdgcn_wmma_f32_16x16x32_bf16(
                false, afrag.v, false, b1.v, (short)0, acc1, false, false);
        }

        // epilogue: pre-activation gates + bias (f32 C/D layout)
        {
            int n0 = nregion + (lane & 15);
            int n1 = n0 + 16;
#pragma unroll
            for (int v = 0; v < 8; ++v) {
                int m = wave * 16 + v + ((lane >> 4) << 3);
                gates[(size_t)m * NGATES + n0] = acc0[v] + bn0;
                gates[(size_t)m * NGATES + n1] = acc1[v] + bn1;
            }
        }

        ++epoch;
        grid_sync(sync_cnt, NBLK * epoch);

        // ---------------- pointwise phase (4 elements per thread) -----------
        {
            const float* xnext = inputs + (size_t)(t + 1) * BATCH * DIN;
            const bool has_next = (t + 1 < T_STEPS);
            const bool is_last  = (t == T_STEPS - 1);
            float* h_out = out + (size_t)t * BATCH * HID;
            int gtid = blockIdx.x * 256 + tid;
#pragma unroll
            for (int r = 0; r < 4; ++r) {
                int idx = gtid + r * (NBLK * 256);     // 0 .. 65535
                int b = idx >> 9, j = idx & 511;
                const float* g = gates + (size_t)b * NGATES;
                float fg = g[j];
                float ig = g[j + 512];
                float gg = g[j + 1024];
                float og = g[j + 1536];
                float f  = 1.0f / (1.0f + __expf(-fg));
                float i  = 1.0f / (1.0f + __expf(-ig));
                float gt = tanhf(gg);
                float o  = 1.0f / (1.0f + __expf(-og));
                float c  = f * c_state[idx] + i * gt;
                float h  = o * tanhf(c);
                c_state[idx] = c;
                h_out[idx]   = h;
                comb[(size_t)b * DCOMB + HID + j] = f32_to_bf16_rne(h);
                if (has_next)
                    comb[(size_t)b * DCOMB + j] = f32_to_bf16_rne(xnext[idx]);
                if (is_last) {
                    out[(size_t)T_STEPS * BATCH * HID + idx] = h;             // hx
                    out[(size_t)T_STEPS * BATCH * HID + BATCH * HID + idx] = c; // cx
                }
            }
        }

        ++epoch;
        grid_sync(sync_cnt, NBLK * epoch);
    }
}

// ---------------------------------------------------------------------------
extern "C" void kernel_launch(void* const* d_in, const int* in_sizes, int n_in,
                              void* d_out, int out_size, void* d_ws, size_t ws_size,
                              hipStream_t stream) {
    (void)in_sizes; (void)n_in; (void)out_size; (void)ws_size;

    const float* inputs = (const float*)d_in[0];
    const float* Wf = (const float*)d_in[1];
    const float* bf = (const float*)d_in[2];
    const float* Wi = (const float*)d_in[3];
    const float* bi = (const float*)d_in[4];
    const float* Wg = (const float*)d_in[5];
    const float* bg = (const float*)d_in[6];
    const float* Wo = (const float*)d_in[7];
    const float* bo = (const float*)d_in[8];
    float* out = (float*)d_out;

    // workspace layout
    uint8_t* ws = (uint8_t*)d_ws;
    size_t off = 0;
    unsigned short* Wsw  = (unsigned short*)(ws + off); off += (size_t)DCOMB * NGATES * 2; // 4 MB
    float*          bias = (float*)(ws + off);          off += (size_t)NGATES * 4;         // 8 KB
    unsigned short* comb = (unsigned short*)(ws + off); off += (size_t)BATCH * DCOMB * 2;  // 256 KB
    float*          cst  = (float*)(ws + off);          off += (size_t)BATCH * HID * 4;    // 256 KB
    float*          gates= (float*)(ws + off);          off += (size_t)BATCH * NGATES * 4; // 1 MB
    unsigned*       scnt = (unsigned*)(ws + off);       off += 256;                        // sync counter

    prep_w_kernel<<<(DCOMB * NGATES + 255) / 256, 256, 0, stream>>>(Wf, Wi, Wg, Wo, Wsw);
    prep_state_kernel<<<(BATCH * HID + 255) / 256, 256, 0, stream>>>(inputs, bf, bi, bg, bo,
                                                                     bias, comb, cst, scnt);

    lstm_persistent_kernel<<<NBLK, 256, 0, stream>>>(inputs, comb, Wsw, bias, gates, cst,
                                                     out, scnt);
}
